// GlobalFeatureMoE_55954833932310
// MI455X (gfx1250) — matrix operand order repaired and checked by reading, not compiled, not run on hardware
//
#include <hip/hip_runtime.h>
#include <stdint.h>

// ---------------- problem constants ----------------
#define B_SZ   4
#define L_SZ   2048
#define T_TOK  8192          // B*L
#define DIM    1024
#define NEXP   8
#define HEXP   4096          // routed & shared hidden

// ---------------- WMMA tile config ----------------
#define BM 128
#define BN 128
#define BK 32
#define LDSP 36              // padded LDS row stride (ushorts): 72B = 18 banks, conflict-free

typedef __attribute__((ext_vector_type(16))) __bf16 v16bf;
typedef __attribute__((ext_vector_type(8)))  float  v8f;

// f32 pair -> packed bf16 dword (RNE) -- force the exact hardware pairing with
// inline asm (the generic-cast path makes clang pair adjacent regs and repack).
__device__ __forceinline__ unsigned int pack2bf(float lo, float hi) {
  unsigned int r;
  asm("v_cvt_pk_bf16_f32 %0, %1, %2" : "=v"(r) : "v"(lo), "v"(hi));
  return r;
}
__device__ __forceinline__ unsigned short f2bf(float f) {
  union { __bf16 b; unsigned short s; } c;
  c.b = (__bf16)f;            // single v_cvt (RNE) on gfx1250
  return c.s;
}
__device__ __forceinline__ __bf16 us2bf(unsigned short s) {
  union { unsigned short s; __bf16 b; } c; c.s = s; return c.b;
}
__device__ __forceinline__ float gelu_tanh(float x) {
  float x3 = x * x * x;
  return 0.5f * x * (1.0f + tanhf(0.7978845608028654f * (x + 0.044715f * x3)));
}

// ---- fragment loaders (CDNA5 16-bit WMMA layouts, wave32) ----
// A 16x32: lanes 0-15 = rows M, lanes 16-31 same rows, interleaved K halves:
//   K = (r>=4?16:0) + (laneHi?8:0) + 2*(r&3) + p
__device__ __forceinline__ v16bf fragA(const unsigned short (*S)[LDSP], int mbase, int lane) {
  int m  = mbase + (lane & 15);
  int kh = (lane >> 4) * 8;
  v16bf f;
#pragma unroll
  for (int r = 0; r < 8; ++r) {
    int kp = ((r & 4) << 2) + kh + ((r & 3) << 1);
    unsigned int w = *(const unsigned int*)&S[m][kp];   // 4B-aligned (kp even, stride 72B)
    f[2 * r]     = us2bf((unsigned short)(w & 0xFFFF));
    f[2 * r + 1] = us2bf((unsigned short)(w >> 16));
  }
  return f;
}
// B 32x16: lanes 0-15 = col N, lanes 16-31 same cols, interleaved K halves (same K order as A).
// Bs is stored [N][K] (transposed) so a column is contiguous in K.
__device__ __forceinline__ v16bf fragB(const unsigned short (*S)[LDSP], int nbase, int lane) {
  int n  = nbase + (lane & 15);
  int kb = (lane >> 4) * 8;
  v16bf f;
#pragma unroll
  for (int r = 0; r < 8; ++r) {
    unsigned int w = *(const unsigned int*)&S[n][((r & 4) << 2) + kb + ((r & 3) << 1)];
    f[2 * r]     = us2bf((unsigned short)(w & 0xFFFF));
    f[2 * r + 1] = us2bf((unsigned short)(w >> 16));
  }
  return f;
}

// ---- double-buffered GEMM mainloop: C(128x128,f32) += A(bf16, row-indexed) * W(f32->bf16) ----
// Pipeline: prefetch tile k+1 into registers while WMMAs consume LDS buffer k,
// then store to the alternate buffer; one barrier per K-step.
__device__ __forceinline__ void wmma_mainloop(
    const unsigned short* __restrict__ A, const float* __restrict__ W,
    const int* __restrict__ tok, int cnt, int m0, int n0, int N, int K,
    unsigned short (*As)[LDSP], unsigned short (*Bs)[LDSP],   // [2*BM][LDSP], [2*BN][LDSP]
    v8f c[2][4])
{
  const int t    = threadIdx.x;
  const int lane = t & 31;
  const int wave = t >> 5;
  const int wm0  = (wave >> 1) * 32;   // 4 wave-rows x 2 wave-cols
  const int wn0  = (wave & 1) * 64;

  // ---- loop-invariant addressing ----
  const int ar = t >> 3;               // A staging: 32 rows x (8 thr * 4 elems)
  const int ac = (t & 7) * 4;
  const unsigned short* aPtr[4];
#pragma unroll
  for (int i = 0; i < 4; ++i) {
    int gr = m0 + ar + i * 32;
    int sr = (gr < cnt) ? gr : (cnt - 1);
    if (tok) sr = tok[sr];             // gather indirection resolved once, outside K loop
    aPtr[i] = A + (size_t)sr * K + ac;
  }
  const int bn = (t & 31) * 4;         // B staging: (2 consecutive k) x (4 consecutive n)
  const int bk = (t >> 5) * 2;
  const float* wPtr = W + n0 + bn;

  uint2  aReg[4];
  float4 bReg[4];

  auto stage_load = [&](int k0) {
#pragma unroll
    for (int i = 0; i < 4; ++i)
      aReg[i] = *(const uint2*)(aPtr[i] + k0);
#pragma unroll
    for (int i = 0; i < 2; ++i) {
      int k = k0 + bk + 16 * i;
      bReg[2 * i]     = *(const float4*)(wPtr + (size_t)k * N);
      bReg[2 * i + 1] = *(const float4*)(wPtr + (size_t)(k + 1) * N);
    }
  };
  auto stage_store = [&](int buf) {
    unsigned short (*Ad)[LDSP] = As + buf * BM;
    unsigned short (*Bd)[LDSP] = Bs + buf * BN;
#pragma unroll
    for (int i = 0; i < 4; ++i) {
      int row = ar + i * 32;
      *(unsigned int*)&Ad[row][ac]     = aReg[i].x;
      *(unsigned int*)&Ad[row][ac + 2] = aReg[i].y;
    }
#pragma unroll
    for (int i = 0; i < 2; ++i) {
      int k = bk + 16 * i;
      float4 r0 = bReg[2 * i], r1 = bReg[2 * i + 1];
      // pack bf16(k) | bf16(k+1) -> one v_cvt_pk_bf16_f32 + one dword DS store per n
      *(unsigned int*)&Bd[bn + 0][k] = pack2bf(r0.x, r1.x);
      *(unsigned int*)&Bd[bn + 1][k] = pack2bf(r0.y, r1.y);
      *(unsigned int*)&Bd[bn + 2][k] = pack2bf(r0.z, r1.z);
      *(unsigned int*)&Bd[bn + 3][k] = pack2bf(r0.w, r1.w);
    }
  };

  stage_load(0);
  stage_store(0);
  __syncthreads();

  for (int k0 = 0; k0 < K; k0 += BK) {
    const int  cur  = (k0 / BK) & 1;
    const bool more = (k0 + BK) < K;
    if (more) stage_load(k0 + BK);     // VMEM for next tile overlaps WMMAs below

    const unsigned short (*Ac)[LDSP] = As + cur * BM;
    const unsigned short (*Bc)[LDSP] = Bs + cur * BN;
    v16bf a[2], b[4];
#pragma unroll
    for (int i = 0; i < 2; ++i) a[i] = fragA(Ac, wm0 + i * 16, lane);
#pragma unroll
    for (int j = 0; j < 4; ++j) b[j] = fragB(Bc, wn0 + j * 16, lane);
#pragma unroll
    for (int i = 0; i < 2; ++i)
#pragma unroll
      for (int j = 0; j < 4; ++j)
        c[i][j] = __builtin_amdgcn_wmma_f32_16x16x32_bf16(
            false, a[i], false, b[j], (short)0, c[i][j], false, false);

    if (more) stage_store(cur ^ 1);
    __syncthreads();
  }
}

// ---- up-projection: H[row] = gelu(A[tok[row]] @ W + bias), bf16 out ----
__global__ __launch_bounds__(256) void k_gemm_up(
    const unsigned short* __restrict__ A, const float* __restrict__ W,
    const float* __restrict__ bias, unsigned short* __restrict__ Hout,
    const int* __restrict__ tok, const int* __restrict__ cntp,
    int rowsMax, int N, int K)
{
  __shared__ unsigned short As[2 * BM][LDSP];
  __shared__ unsigned short Bs[2 * BN][LDSP];
  const int cnt = cntp ? *cntp : rowsMax;
  const int m0  = blockIdx.y * BM;
  if (m0 >= cnt) return;
  const int n0 = blockIdx.x * BN;

  v8f c[2][4];
  const v8f z = {0.f, 0.f, 0.f, 0.f, 0.f, 0.f, 0.f, 0.f};
#pragma unroll
  for (int i = 0; i < 2; ++i)
#pragma unroll
    for (int j = 0; j < 4; ++j) c[i][j] = z;

  wmma_mainloop(A, W, tok, cnt, m0, n0, N, K, As, Bs, c);

  const int lane = threadIdx.x & 31;
  const int wave = threadIdx.x >> 5;
  const int wm0 = (wave >> 1) * 32, wn0 = (wave & 1) * 64;
  const int l15 = lane & 15, mh = (lane >> 4) * 8;
#pragma unroll
  for (int i = 0; i < 2; ++i)
#pragma unroll
    for (int r = 0; r < 8; ++r) {
      const int m = m0 + wm0 + i * 16 + r + mh;
      if (m < cnt) {
        unsigned short* hr = Hout + (size_t)m * N;
#pragma unroll
        for (int j = 0; j < 4; ++j) {
          const int n = n0 + wn0 + j * 16 + l15;
          hr[n] = f2bf(gelu_tanh(c[i][j][r] + bias[n]));
        }
      }
    }
}

// ---- down-projection: Y[tok[row]] (=| +=) wgt[row] * (A[tok[row]] @ W + bias), f32 out ----
__global__ __launch_bounds__(256) void k_gemm_down(
    const unsigned short* __restrict__ A, const float* __restrict__ W,
    const float* __restrict__ bias, float* __restrict__ Y,
    const int* __restrict__ tok, const float* __restrict__ wgt,
    const int* __restrict__ cntp, int rowsMax, int N, int K, int accum)
{
  __shared__ unsigned short As[2 * BM][LDSP];
  __shared__ unsigned short Bs[2 * BN][LDSP];
  const int cnt = cntp ? *cntp : rowsMax;
  const int m0  = blockIdx.y * BM;
  if (m0 >= cnt) return;
  const int n0 = blockIdx.x * BN;

  v8f c[2][4];
  const v8f z = {0.f, 0.f, 0.f, 0.f, 0.f, 0.f, 0.f, 0.f};
#pragma unroll
  for (int i = 0; i < 2; ++i)
#pragma unroll
    for (int j = 0; j < 4; ++j) c[i][j] = z;

  wmma_mainloop(A, W, tok, cnt, m0, n0, N, K, As, Bs, c);

  const int lane = threadIdx.x & 31;
  const int wave = threadIdx.x >> 5;
  const int wm0 = (wave >> 1) * 32, wn0 = (wave & 1) * 64;
  const int l15 = lane & 15, mh = (lane >> 4) * 8;
#pragma unroll
  for (int i = 0; i < 2; ++i)
#pragma unroll
    for (int r = 0; r < 8; ++r) {
      const int m = m0 + wm0 + i * 16 + r + mh;
      if (m < cnt) {
        const int   orow = tok ? tok[m] : m;      // hoisted: one load per output row
        const float s    = wgt ? wgt[m] : 1.0f;
        float* yr = Y + (size_t)orow * N;
#pragma unroll
        for (int j = 0; j < 4; ++j) {
          const int n = n0 + wn0 + j * 16 + l15;
          float v = s * (c[i][j][r] + bias[n]);
          if (accum) yr[n] += v; else yr[n] = v;  // rows unique per expert launch
        }
      }
    }
}

// ---- cond = silu(time_c) @ ada_w + ada_b  (B x 2D) ----
__global__ __launch_bounds__(256) void k_cond(
    const float* __restrict__ tc, const float* __restrict__ aw,
    const float* __restrict__ ab, float* __restrict__ cond)
{
  int idx = blockIdx.x * 256 + threadIdx.x;       // < B*2D = 8192
  int b = idx >> 11, j = idx & 2047;
  float acc = ab[j];
  for (int d = 0; d < DIM; ++d) {
    float v = tc[b * DIM + d];
    float s = v / (1.0f + expf(-v));
    acc += s * aw[(size_t)d * (2 * DIM) + j];
  }
  cond[idx] = acc;
}

// ---- fused adaLN + router (one block per token) ----
__global__ __launch_bounds__(256) void k_ln_router(
    const float* __restrict__ x, const float* __restrict__ cond,
    const float* __restrict__ gate_w, unsigned short* __restrict__ xn,
    int* __restrict__ topi, float* __restrict__ topw)
{
  __shared__ float rs[256], rs2[256];
  __shared__ float pg[256][NEXP];
  const int tkn = blockIdx.x;
  const int b   = tkn >> 11;                     // L = 2048
  const int tid = threadIdx.x;
  const float* xr = x + (size_t)tkn * DIM;

  float s = 0.f, s2 = 0.f;
#pragma unroll
  for (int i = 0; i < 4; ++i) { float v = xr[tid + i * 256]; s += v; s2 += v * v; }
  rs[tid] = s; rs2[tid] = s2; __syncthreads();
  for (int st = 128; st > 0; st >>= 1) {
    if (tid < st) { rs[tid] += rs[tid + st]; rs2[tid] += rs2[tid + st]; }
    __syncthreads();
  }
  const float mean = rs[0] * (1.0f / DIM);
  const float var  = rs2[0] * (1.0f / DIM) - mean * mean;
  const float inv  = rsqrtf(var + 1e-6f);

  const float* shift = cond + (size_t)b * (2 * DIM);
  const float* scale = shift + DIM;
  float acc[NEXP];
#pragma unroll
  for (int e = 0; e < NEXP; ++e) acc[e] = 0.f;
#pragma unroll
  for (int i = 0; i < 4; ++i) {
    int d = tid + i * 256;
    float v = (xr[d] - mean) * inv;
    v = v * (1.0f + scale[d]) + shift[d];
    xn[(size_t)tkn * DIM + d] = f2bf(v);
    const float* g = gate_w + (size_t)d * NEXP;
#pragma unroll
    for (int e = 0; e < NEXP; ++e) acc[e] += v * g[e];
  }
#pragma unroll
  for (int e = 0; e < NEXP; ++e) pg[tid][e] = acc[e];
  __syncthreads();
  for (int st = 128; st > 0; st >>= 1) {
    if (tid < st)
#pragma unroll
      for (int e = 0; e < NEXP; ++e) pg[tid][e] += pg[tid + st][e];
    __syncthreads();
  }
  if (tid == 0) {
    float l[NEXP], mx = -1e30f;
#pragma unroll
    for (int e = 0; e < NEXP; ++e) { l[e] = pg[0][e]; mx = fmaxf(mx, l[e]); }
    float sum = 0.f;
#pragma unroll
    for (int e = 0; e < NEXP; ++e) { l[e] = expf(l[e] - mx); sum += l[e]; }
    const float isum = 1.0f / sum;
    int i1 = 0;
    for (int e = 1; e < NEXP; ++e) if (l[e] > l[i1]) i1 = e;
    int i2 = (i1 == 0) ? 1 : 0;
    for (int e = 0; e < NEXP; ++e) if (e != i1 && l[e] > l[i2]) i2 = e;
    const float v1 = l[i1] * isum, v2 = l[i2] * isum;
    const float rn = 1.0f / (v1 + v2);
    topi[2 * tkn] = i1; topi[2 * tkn + 1] = i2;
    topw[2 * tkn] = v1 * rn; topw[2 * tkn + 1] = v2 * rn;
  }
}

__global__ void k_zero_counts(int* __restrict__ counts) {
  if (threadIdx.x < NEXP) counts[threadIdx.x] = 0;
}

__global__ __launch_bounds__(256) void k_gather(
    const int* __restrict__ topi, const float* __restrict__ topw,
    int* __restrict__ counts, int* __restrict__ tokl, float* __restrict__ wgt)
{
  int t = blockIdx.x * 256 + threadIdx.x;
  if (t >= T_TOK) return;
#pragma unroll
  for (int k = 0; k < 2; ++k) {
    int e = topi[2 * t + k];
    int p = atomicAdd(&counts[e], 1);
    tokl[(size_t)e * T_TOK + p] = t;
    wgt[(size_t)e * T_TOK + p]  = topw[2 * t + k];
  }
}

__global__ __launch_bounds__(256) void k_finalize(
    const float* __restrict__ x, const float* __restrict__ y, float* __restrict__ out)
{
  size_t i = (size_t)blockIdx.x * 256 + threadIdx.x;
  if (i < (size_t)T_TOK * DIM) out[i] = x[i] + y[i];
}

extern "C" void kernel_launch(void* const* d_in, const int* in_sizes, int n_in,
                              void* d_out, int out_size, void* d_ws, size_t ws_size,
                              hipStream_t stream) {
  const float* x      = (const float*)d_in[0];
  const float* time_c = (const float*)d_in[1];
  const float* ada_w  = (const float*)d_in[2];
  const float* ada_b  = (const float*)d_in[3];
  const float* gate_w = (const float*)d_in[4];
  const float* w1     = (const float*)d_in[5];
  const float* b1     = (const float*)d_in[6];
  const float* w2     = (const float*)d_in[7];
  const float* b2     = (const float*)d_in[8];
  const float* sw1    = (const float*)d_in[9];
  const float* sb1    = (const float*)d_in[10];
  const float* sw2    = (const float*)d_in[11];
  const float* sb2    = (const float*)d_in[12];
  float* out = (float*)d_out;
  (void)in_sizes; (void)n_in; (void)out_size; (void)ws_size;

  // workspace layout (all offsets multiples of 256B)
  char* w = (char*)d_ws;
  size_t off = 0;
  float*          cond = (float*)(w + off);          off += (size_t)B_SZ * 2 * DIM * 4;     // 32 KB
  unsigned short* xn   = (unsigned short*)(w + off); off += (size_t)T_TOK * DIM * 2;        // 16 MB
  int*            topi = (int*)(w + off);            off += (size_t)T_TOK * 2 * 4;
  float*          topw = (float*)(w + off);          off += (size_t)T_TOK * 2 * 4;
  int*            cnts = (int*)(w + off);            off += 256;
  int*            tokl = (int*)(w + off);            off += (size_t)NEXP * T_TOK * 4;
  float*          wgt  = (float*)(w + off);          off += (size_t)NEXP * T_TOK * 4;
  unsigned short* hbuf = (unsigned short*)(w + off); off += (size_t)T_TOK * HEXP * 2;       // 64 MB (shared & experts reuse)
  float*          ybuf = (float*)(w + off);          off += (size_t)T_TOK * DIM * 4;        // 32 MB

  dim3 blk(256);
  k_zero_counts<<<1, 32, 0, stream>>>(cnts);
  k_cond<<<(B_SZ * 2 * DIM) / 256, blk, 0, stream>>>(time_c, ada_w, ada_b, cond);
  k_ln_router<<<T_TOK, blk, 0, stream>>>(x, cond, gate_w, xn, topi, topw);
  k_gather<<<T_TOK / 256, blk, 0, stream>>>(topi, topw, cnts, tokl, wgt);

  dim3 gu(HEXP / BN, T_TOK / BM);   // 32 x 64
  dim3 gd(DIM / BN,  T_TOK / BM);   //  8 x 64

  // shared expert (all tokens): y = (gelu(xn@sw1+sb1))@sw2 + sb2
  k_gemm_up  <<<gu, blk, 0, stream>>>(xn, sw1, sb1, hbuf, nullptr, nullptr, T_TOK, HEXP, DIM);
  k_gemm_down<<<gd, blk, 0, stream>>>(hbuf, sw2, sb2, ybuf, nullptr, nullptr, nullptr,
                                      T_TOK, DIM, HEXP, /*accum=*/0);

  // routed experts: gathered grouped-GEMMs, serialized on stream (hbuf reuse is safe)
  for (int e = 0; e < NEXP; ++e) {
    k_gemm_up  <<<gu, blk, 0, stream>>>(xn, w1 + (size_t)e * DIM * HEXP, b1 + (size_t)e * HEXP,
                                        hbuf, tokl + (size_t)e * T_TOK, cnts + e,
                                        T_TOK, HEXP, DIM);
    k_gemm_down<<<gd, blk, 0, stream>>>(hbuf, w2 + (size_t)e * HEXP * DIM, b2 + (size_t)e * DIM,
                                        ybuf, tokl + (size_t)e * T_TOK, wgt + (size_t)e * T_TOK,
                                        cnts + e, T_TOK, DIM, HEXP, /*accum=*/1);
  }

  k_finalize<<<(T_TOK * DIM) / 256, blk, 0, stream>>>(x, ybuf, out);
}
